// STEBitLinear_46437186404373
// MI455X (gfx1250) — matrix operand order repaired and checked by reading, not compiled, not run on hardware
//
#include <hip/hip_runtime.h>

#define IN_F   4096
#define OUT_F  4096
#define M_TOT  8192
#define GROUP  128

#define BM 256
#define BN 128
#define BK 32
#define LDT 40                 // padded LDS row stride in bf16 elems (80B, 16B-aligned)
#define NK (IN_F / BK)         // 128 K-steps

typedef __attribute__((ext_vector_type(16))) __bf16 v16bf;
typedef __attribute__((ext_vector_type(8)))  __bf16 v8bf;
typedef __attribute__((ext_vector_type(4)))  __bf16 v4bf;
typedef __attribute__((ext_vector_type(8)))  float  v8f;
typedef __attribute__((ext_vector_type(4)))  float  v4f;

__global__ __launch_bounds__(256)
void ste_bitlinear_wmma(const float* __restrict__ X,   // [M_TOT][IN_F]
                        const float* __restrict__ W,   // [OUT_F][IN_F], values +-1
                        const float* __restrict__ S,   // [OUT_F*IN_F/GROUP]
                        float* __restrict__ O)         // [M_TOT][OUT_F]
{
    __shared__ __bf16 As[2][BM * LDT];   // 256 x 32 bf16, padded
    __shared__ __bf16 Bs[2][BN * LDT];   // 128 x 32 bf16, padded

    const int tid  = threadIdx.x;
    const int lane = tid & 31;
    const int wid  = tid >> 5;        // 0..7
    const int wm   = wid >> 1;        // 0..3 : 64-row slab
    const int wn   = wid & 1;         // 0..1 : 64-col slab

    const int m0 = blockIdx.y * BM;
    const int n0 = blockIdx.x * BN;

    // global->LDS staging map: thread covers float4 col gc/4, rows gr+32*i
    const int gc = (tid & 7) * 4;     // k offset within tile: 0..28
    const int gr = tid >> 3;          // 0..31

    v8f acc[4][4] = {};

    v4f ax[8];        // A staging: 8 rows (256 rows / 32)
    v4f bw[4];        // B staging: 4 rows (128 rows / 32)
    float sc[4];

    // ---------------- prologue: stage K-tile 0 ----------------
    {
        const int k0 = 0;
#pragma unroll
        for (int i = 0; i < 8; ++i)
            ax[i] = *(const v4f*)(X + (size_t)(m0 + gr + 32 * i) * IN_F + k0 + gc);
#pragma unroll
        for (int i = 0; i < 4; ++i) {
            const int r = gr + 32 * i;
            bw[i] = *(const v4f*)(W + (size_t)(n0 + r) * IN_F + k0 + gc);
            sc[i] = S[(n0 + r) * (IN_F / GROUP) + (k0 >> 7)];
        }
#pragma unroll
        for (int i = 0; i < 8; ++i) {
            v4bf ha;
#pragma unroll
            for (int e = 0; e < 4; ++e) ha[e] = (__bf16)ax[i][e];
            *(v4bf*)&As[0][(gr + 32 * i) * LDT + gc] = ha;
        }
#pragma unroll
        for (int i = 0; i < 4; ++i) {
            v4bf hb;
#pragma unroll
            for (int e = 0; e < 4; ++e) hb[e] = (__bf16)(bw[i][e] * sc[i]);
            *(v4bf*)&Bs[0][(gr + 32 * i) * LDT + gc] = hb;
        }
    }
    __syncthreads();

    // per-lane fragment addressing (wave32 WMMA VGPR layouts)
    const int a_k0 = (lane & 16) ? 8 : 0;    // A: halves 0-7 = K0..7 / K8..15, +16 for hi
    const int b_k0 = (lane & 16) ? 8 : 0;    // B: halves 0-7 = K0..7 / K8..15, +16 for hi
    const int a_row = wm * 64 + (lane & 15);
    const int b_col = wn * 64 + (lane & 15);

    for (int kt = 0; kt < NK; ++kt) {
        const int cur  = kt & 1;
        const int nxt  = cur ^ 1;
        const bool more = (kt + 1) < NK;

        if (more) {
            const int k0 = (kt + 1) * BK;
#pragma unroll
            for (int i = 0; i < 8; ++i)
                ax[i] = *(const v4f*)(X + (size_t)(m0 + gr + 32 * i) * IN_F + k0 + gc);
#pragma unroll
            for (int i = 0; i < 4; ++i) {
                const int r = gr + 32 * i;
                bw[i] = *(const v4f*)(W + (size_t)(n0 + r) * IN_F + k0 + gc);
                sc[i] = S[(n0 + r) * (IN_F / GROUP) + (k0 >> 7)];
            }
        }
        if (kt + 2 < NK) {
            // pull tile kt+2 toward L2 (global_prefetch_b8)
            __builtin_prefetch(X + (size_t)(m0 + gr) * IN_F + (kt + 2) * BK + gc, 0, 1);
            __builtin_prefetch(W + (size_t)(n0 + gr) * IN_F + (kt + 2) * BK + gc, 0, 1);
        }

        // ---------------- compute on `cur` buffer ----------------
        v16bf afrag[4], bfrag[4];
#pragma unroll
        for (int i = 0; i < 4; ++i) {
            const __bf16* p = &As[cur][(a_row + i * 16) * LDT + a_k0];
            v8bf lo = *(const v8bf*)(p);        // K 0..7   (or 8..15)
            v8bf hi = *(const v8bf*)(p + 16);   // K 16..23 (or 24..31)
#pragma unroll
            for (int e = 0; e < 8; ++e) { afrag[i][e] = lo[e]; afrag[i][e + 8] = hi[e]; }
        }
#pragma unroll
        for (int j = 0; j < 4; ++j) {
            const __bf16* p = &Bs[cur][(b_col + j * 16) * LDT + b_k0];
            v8bf lo = *(const v8bf*)(p);
            v8bf hi = *(const v8bf*)(p + 16);
#pragma unroll
            for (int e = 0; e < 8; ++e) { bfrag[j][e] = lo[e]; bfrag[j][e + 8] = hi[e]; }
        }
#pragma unroll
        for (int i = 0; i < 4; ++i)
#pragma unroll
            for (int j = 0; j < 4; ++j)
                acc[i][j] = __builtin_amdgcn_wmma_f32_16x16x32_bf16(
                    false, afrag[i], false, bfrag[j],
                    (short)0, acc[i][j], false, false);

        // ---------------- stage K-tile kt+1 into `nxt` ----------------
        if (more) {
#pragma unroll
            for (int i = 0; i < 8; ++i) {
                v4bf ha;
#pragma unroll
                for (int e = 0; e < 4; ++e) ha[e] = (__bf16)ax[i][e];
                *(v4bf*)&As[nxt][(gr + 32 * i) * LDT + gc] = ha;
            }
#pragma unroll
            for (int i = 0; i < 4; ++i) {
                v4bf hb;
#pragma unroll
                for (int e = 0; e < 4; ++e) hb[e] = (__bf16)(bw[i][e] * sc[i]);
                *(v4bf*)&Bs[nxt][(gr + 32 * i) * LDT + gc] = hb;
            }
        }
        __syncthreads();
    }

    // ---------------- epilogue: C/D layout -> global ----------------
    // lanes 0-15: M = vgpr idx; lanes 16-31: M = vgpr idx + 8; N = lane & 15
    const int orow0 = m0 + wm * 64 + ((lane >> 4) * 8);
    const int ocol  = n0 + wn * 64 + (lane & 15);
#pragma unroll
    for (int i = 0; i < 4; ++i)
#pragma unroll
        for (int j = 0; j < 4; ++j)
#pragma unroll
            for (int v = 0; v < 8; ++v) {
                const int r = orow0 + i * 16 + v;
                O[(size_t)r * OUT_F + ocol + j * 16] = acc[i][j][v];
            }
}

extern "C" void kernel_launch(void* const* d_in, const int* in_sizes, int n_in,
                              void* d_out, int out_size, void* d_ws, size_t ws_size,
                              hipStream_t stream) {
    (void)in_sizes; (void)n_in; (void)d_ws; (void)ws_size; (void)out_size;
    const float* x  = (const float*)d_in[0];   // (4, 2048, 4096) f32 -> 8192 x 4096
    const float* sw = (const float*)d_in[1];   // (4096, 4096) f32, +-1
    const float* sc = (const float*)d_in[2];   // (131072,) f32 per-group scales
    float* out = (float*)d_out;                // (4, 2048, 4096) f32

    dim3 grid(OUT_F / BN, M_TOT / BM);         // (32, 32)
    dim3 block(256);
    ste_bitlinear_wmma<<<grid, block, 0, stream>>>(x, sw, sc, out);
}
